// Attention_29008209117187
// MI455X (gfx1250) — matrix operand order repaired and checked by reading, not compiled, not run on hardware
//
#include <hip/hip_runtime.h>
#include <hip/hip_bf16.h>

typedef __attribute__((ext_vector_type(16))) __bf16 bf16x16;
typedef __attribute__((ext_vector_type(8)))  __bf16 bf16x8;
typedef __attribute__((ext_vector_type(8)))  float  floatx8;

// ---------------------------------------------------------------------------
// WMMA + fragment helpers (CDNA5 v_wmma_f32_16x16x32_bf16, wave32)
// ---------------------------------------------------------------------------
__device__ __forceinline__ floatx8 wmma_bf16(bf16x16 a, bf16x16 b, floatx8 c) {
  return __builtin_amdgcn_wmma_f32_16x16x32_bf16(
      false, a, false, b, (short)0, c, false, false);
}

__device__ __forceinline__ bf16x16 join16(bf16x8 lo, bf16x8 hi) {
  bf16x16 r;
#pragma unroll
  for (int e = 0; e < 8; ++e) { r[e] = lo[e]; r[e + 8] = hi[e]; }
  return r;
}

// A fragment from row-major [row][K] tile: K pattern {half*8+0..7, 16+half*8+0..7}
// -> two contiguous 16B LDS loads.
__device__ __forceinline__ bf16x16 a_frag(const __bf16* tile, int ld, int lane) {
  const int row = lane & 15, half = lane >> 4;
  const __bf16* p = tile + row * ld + half * 8;
  return join16(*(const bf16x8*)p, *(const bf16x8*)(p + 16));
}

// B fragment from a [col][K] tile (B^T row-major): K pattern {half*8+0..7, 16+half*8+0..7}
// -> two contiguous 16B LDS loads.
__device__ __forceinline__ bf16x16 bT_frag(const __bf16* tile, int ld, int lane) {
  const int col = lane & 15, half = lane >> 4;
  const __bf16* p = tile + col * ld + half * 8;
  return join16(*(const bf16x8*)p, *(const bf16x8*)(p + 16));
}

// 16 fp32 (float4 x4, coalesced) -> 16 bf16 in LDS (2 x 16B stores)
__device__ __forceinline__ void stage16(const float* __restrict__ src,
                                        __bf16* __restrict__ dst) {
  const float4* s4 = (const float4*)src;
  float v[16];
#pragma unroll
  for (int j = 0; j < 4; ++j) {
    float4 q = s4[j];
    v[4 * j] = q.x; v[4 * j + 1] = q.y; v[4 * j + 2] = q.z; v[4 * j + 3] = q.w;
  }
  bf16x8 lo, hi;
#pragma unroll
  for (int j = 0; j < 8; ++j) { lo[j] = (__bf16)v[j]; hi[j] = (__bf16)v[8 + j]; }
  *(bf16x8*)dst = lo;
  *(bf16x8*)(dst + 8) = hi;
}

// ---------------------------------------------------------------------------
// Kernel 1: out^T orientation: A = W rows (c), B^T = x rows (m).
// No transposes anywhere; epilogue = 1 x b128 store per tile.
// z=0 -> q, z=1 -> kv (k == v in the reference!).
// ---------------------------------------------------------------------------
__global__ __launch_bounds__(128) void gemm_qkv(
    const float* __restrict__ x, const float* __restrict__ Wq,
    const float* __restrict__ Wkv, __bf16* __restrict__ qbf,
    __bf16* __restrict__ kvbf) {
  __shared__ __align__(16) __bf16 Ws[64 * 32];
  __shared__ __align__(16) __bf16 Xs[64 * 32];
  const int m0 = blockIdx.x * 64, c0 = blockIdx.y * 64;
  const float* W = blockIdx.z ? Wkv : Wq;
  __bf16* dst = blockIdx.z ? kvbf : qbf;
  const int tid = threadIdx.x, lane = tid & 31, wvid = tid >> 5;
  const int srow = tid >> 1, skh = (tid & 1) * 16;
  floatx8 acc[4] = {};
  for (int k0 = 0; k0 < 512; k0 += 32) {
    stage16(&W[(size_t)(c0 + srow) * 512 + k0 + skh], &Ws[srow * 32 + skh]);
    stage16(&x[(size_t)(m0 + srow) * 512 + k0 + skh], &Xs[srow * 32 + skh]);
    __syncthreads();
    bf16x16 a = a_frag(&Ws[(wvid * 16) * 32], 32, lane);
#pragma unroll
    for (int t = 0; t < 4; ++t)
      acc[t] = wmma_bf16(a, bT_frag(&Xs[(t * 16) * 32], 32, lane), acc[t]);
    __syncthreads();
  }
  const int col = lane & 15, half = lane >> 4;
#pragma unroll
  for (int t = 0; t < 4; ++t) {                 // lane: (c-stripe rows, m col)
    bf16x8 o;
#pragma unroll
    for (int i = 0; i < 8; ++i) o[i] = (__bf16)acc[t][i];
    int m = m0 + t * 16 + col;
    *(bf16x8*)&dst[(size_t)m * 512 + c0 + wvid * 16 + half * 8] = o;
  }
}

// ---------------------------------------------------------------------------
// Kernel 2: kvT[b][c][n] = kv[b][n][c]  (channel-major copy for V^T operands)
// ---------------------------------------------------------------------------
__global__ __launch_bounds__(128) void transpose_kv(
    const __bf16* __restrict__ kvbf, __bf16* __restrict__ kvT) {
  __shared__ __align__(16) __bf16 tile[64 * 72];   // pad 72 vs bank conflicts
  const int n0 = blockIdx.x * 64, c0 = blockIdx.y * 64, b = blockIdx.z;
  const int tid = threadIdx.x;
#pragma unroll
  for (int i = 0; i < 4; ++i) {
    int ch = tid + i * 128, r = ch >> 3, ec = (ch & 7) * 8;
    *(bf16x8*)&tile[r * 72 + ec] =
        *(const bf16x8*)&kvbf[(size_t)(b * 8192 + n0 + r) * 512 + c0 + ec];
  }
  __syncthreads();
#pragma unroll
  for (int i = 0; i < 4; ++i) {
    int ch = tid + i * 128, e = ch >> 3, rc = (ch & 7) * 8;
    bf16x8 o;
#pragma unroll
    for (int j = 0; j < 8; ++j) o[j] = tile[(rc + j) * 72 + e];
    *(bf16x8*)&kvT[((size_t)b * 512 + c0 + e) * 8192 + n0 + rc] = o;
  }
}

// ---------------------------------------------------------------------------
// Kernel 3: local windowed attention, heads 0..3.
// Scores (natural) -> softmax -> O^T = V^T @ P^T (all fragments contiguous).
// kvT tile overlays the dead qs+kvs region: 56KB LDS total.
// ---------------------------------------------------------------------------
__global__ __launch_bounds__(128) void local_attn(
    const __bf16* __restrict__ qbf, const __bf16* __restrict__ kvbf,
    const __bf16* __restrict__ kvT_g, float* __restrict__ attn_out) {
  __shared__ __align__(16) __bf16 smem[28672];  // 56KB
  __bf16* qs  = smem;           // [64][64]   (phase 1)
  __bf16* kvs = smem + 4096;    // [192][64]  (phase 1)
  __bf16* kvT = smem;           // [64][192]  (phase 2 overlay)
  __bf16* ps  = smem + 16384;   // [64][192]
  const int w = blockIdx.x, h = blockIdx.y, b = blockIdx.z;
  const int tid = threadIdx.x, lane = tid & 31, wvid = tid >> 5;
#pragma unroll
  for (int i = 0; i < 4; ++i) {                 // qs copy (512 x 8-elem chunks)
    int ch = tid + i * 128, r = ch >> 3, ec = (ch & 7) * 8;
    *(bf16x8*)&qs[r * 64 + ec] =
        *(const bf16x8*)&qbf[(size_t)(b * 8192 + w * 64 + r) * 512 + h * 64 + ec];
  }
#pragma unroll
  for (int i = 0; i < 12; ++i) {                // kv window (zero-fill edges)
    int ch = tid + i * 128, j = ch >> 3, ec = (ch & 7) * 8;
    int pos = (w - 1) * 64 + j;
    bf16x8 v = {};
    if (pos >= 0 && pos < 8192)
      v = *(const bf16x8*)&kvbf[(size_t)(b * 8192 + pos) * 512 + h * 64 + ec];
    *(bf16x8*)&kvs[j * 64 + ec] = v;
  }
  __syncthreads();
  floatx8 s[12] = {};                           // 16 q-rows x 192 keys / wave
#pragma unroll
  for (int d0 = 0; d0 < 64; d0 += 32) {
    bf16x16 a = a_frag(&qs[(wvid * 16) * 64 + d0], 64, lane);
#pragma unroll
    for (int t = 0; t < 12; ++t)
      s[t] = wmma_bf16(a, bT_frag(&kvs[(t * 16) * 64 + d0], 64, lane), s[t]);
  }
  const int col = lane & 15, half = lane >> 4;
#pragma unroll
  for (int t = 0; t < 12; ++t) {                // scale + mask
    int pos = (w - 1) * 64 + t * 16 + col;
    bool valid = (pos >= 0 && pos < 8192);
#pragma unroll
    for (int i = 0; i < 8; ++i)
      s[t][i] = valid ? 0.125f * s[t][i] : -1.0e9f;
  }
#pragma unroll
  for (int i = 0; i < 8; ++i) {                 // row softmax (16-lane halves)
    float mx = -3.4e38f;
#pragma unroll
    for (int t = 0; t < 12; ++t) mx = fmaxf(mx, s[t][i]);
#pragma unroll
    for (int m = 1; m <= 8; m <<= 1) mx = fmaxf(mx, __shfl_xor(mx, m, 32));
    float sum = 0.0f;
#pragma unroll
    for (int t = 0; t < 12; ++t) { s[t][i] = __expf(s[t][i] - mx); sum += s[t][i]; }
#pragma unroll
    for (int m = 1; m <= 8; m <<= 1) sum += __shfl_xor(sum, m, 32);
    float inv = 1.0f / sum;
#pragma unroll
    for (int t = 0; t < 12; ++t)
      ps[(wvid * 16 + half * 8 + i) * 192 + t * 16 + col] = (__bf16)(s[t][i] * inv);
  }
  __syncthreads();                              // qs/kvs dead -> overlay kvT
#pragma unroll
  for (int i = 0; i < 12; ++i) {                // kvT tile [e][192 keys]
    int ch = tid + i * 128, e = ch / 24, jc = (ch % 24) * 8;
    int pos = (w - 1) * 64 + jc;
    bf16x8 v = {};
    if (pos >= 0 && pos < 8192)
      v = *(const bf16x8*)&kvT_g[((size_t)b * 512 + h * 64 + e) * 8192 + pos];
    *(bf16x8*)&kvT[e * 192 + jc] = v;
  }
  __syncthreads();
  floatx8 o[4] = {};                            // O^T = V^T @ P^T
#pragma unroll
  for (int k0 = 0; k0 < 192; k0 += 32) {
    bf16x16 a = a_frag(&kvT[(wvid * 16) * 192 + k0], 192, lane);
#pragma unroll
    for (int t = 0; t < 4; ++t)
      o[t] = wmma_bf16(a, bT_frag(&ps[(t * 16) * 192 + k0], 192, lane), o[t]);
  }
#pragma unroll
  for (int t = 0; t < 4; ++t) {                 // lane: (e-stripe, n col)
    int n = w * 64 + t * 16 + col;
    float* dstp =
        &attn_out[((size_t)(b * 8 + h) * 8192 + n) * 64 + wvid * 16 + half * 8];
    *(float4*)dstp       = make_float4(o[t][0], o[t][1], o[t][2], o[t][3]);
    *((float4*)dstp + 1) = make_float4(o[t][4], o[t][5], o[t][6], o[t][7]);
  }
}

// ---------------------------------------------------------------------------
// Kernel 4: column sums of exp(k) over N (k-softmax denominators, heads 4..7)
// ---------------------------------------------------------------------------
__global__ __launch_bounds__(256) void ksum_kernel(
    const __bf16* __restrict__ kvbf, float* __restrict__ ksum) {
  __shared__ float red[4][64];
  const int bh = blockIdx.x, b = bh >> 2, h = (bh & 3) + 4;
  const int d = threadIdx.x & 63, g = threadIdx.x >> 6;
  float s = 0.0f;
  for (int n = g; n < 8192; n += 4)
    s += __expf((float)kvbf[(size_t)(b * 8192 + n) * 512 + h * 64 + d]);
  red[g][d] = s;
  __syncthreads();
  if (g == 0) ksum[bh * 64 + d] = red[0][d] + red[1][d] + red[2][d] + red[3][d];
}

// ---------------------------------------------------------------------------
// Kernel 5: ctx^T[e][d] = (exp(K)^T @ V)^T / ksum[d].  Both tiles are straight
// contiguous copies of the same kvT slice (exp applied elementwise on stage).
// ---------------------------------------------------------------------------
__global__ __launch_bounds__(128) void ctx_kernel(
    const __bf16* __restrict__ kvT_g, const float* __restrict__ ksum,
    __bf16* __restrict__ ctxT) {
  __shared__ __align__(16) __bf16 ek[64 * 32];   // exp(k): rows d
  __shared__ __align__(16) __bf16 vt[64 * 32];   // v:      rows e
  const int bh = blockIdx.x, b = bh >> 2, h = (bh & 3) + 4;
  const int tid = threadIdx.x, lane = tid & 31, wvid = tid >> 5;
  const int srow = tid >> 1, snh = (tid & 1) * 16;
  const size_t base = ((size_t)b * 512 + h * 64 + srow) * 8192;
  floatx8 acc[4] = {};
  for (int n0 = 0; n0 < 8192; n0 += 32) {
    bf16x8 r0 = *(const bf16x8*)&kvT_g[base + n0 + snh];
    bf16x8 r1 = *(const bf16x8*)&kvT_g[base + n0 + snh + 8];
    *(bf16x8*)&vt[srow * 32 + snh] = r0;
    *(bf16x8*)&vt[srow * 32 + snh + 8] = r1;
    bf16x8 e0, e1;
#pragma unroll
    for (int j = 0; j < 8; ++j) {
      e0[j] = (__bf16)__expf((float)r0[j]);
      e1[j] = (__bf16)__expf((float)r1[j]);
    }
    *(bf16x8*)&ek[srow * 32 + snh] = e0;
    *(bf16x8*)&ek[srow * 32 + snh + 8] = e1;
    __syncthreads();
    bf16x16 a = a_frag(&ek[(wvid * 16) * 32], 32, lane);   // rows d
#pragma unroll
    for (int t = 0; t < 4; ++t)
      acc[t] = wmma_bf16(a, bT_frag(&vt[(t * 16) * 32], 32, lane), acc[t]);
    __syncthreads();
  }
  const int col = lane & 15, half = lane >> 4;
  float inv[8];
#pragma unroll
  for (int i = 0; i < 8; ++i)
    inv[i] = 1.0f / ksum[bh * 64 + wvid * 16 + half * 8 + i];
#pragma unroll
  for (int t = 0; t < 4; ++t) {                 // lane: (d-stripe, e col)
    bf16x8 ov;
#pragma unroll
    for (int i = 0; i < 8; ++i) ov[i] = (__bf16)(acc[t][i] * inv[i]);
    *(bf16x8*)&ctxT[(size_t)bh * 4096 + (t * 16 + col) * 64 + wvid * 16 + half * 8] = ov;
  }
}

// ---------------------------------------------------------------------------
// Kernel 6: O^T = ctx^T @ q_sm^T  (heads 4..7).  A = ctxT rows e (contiguous),
// B^T = q_sm rows n (contiguous), output packs over e -> float4 stores.
// ---------------------------------------------------------------------------
__global__ __launch_bounds__(128) void linear_out(
    const __bf16* __restrict__ qbf, const __bf16* __restrict__ ctxT,
    float* __restrict__ attn_out) {
  __shared__ __align__(16) __bf16 qs[64 * 64];
  __shared__ __align__(16) __bf16 cb[64 * 64];
  const int n0 = blockIdx.x * 64, gh = blockIdx.y, b = blockIdx.z, h = gh + 4;
  const int tid = threadIdx.x, lane = tid & 31, wvid = tid >> 5;
#pragma unroll
  for (int i = 0; i < 4; ++i) {                 // ctxT copy
    int off = (tid + i * 128) * 8;
    *(bf16x8*)&cb[off] = *(const bf16x8*)&ctxT[(size_t)(b * 4 + gh) * 4096 + off];
  }
  if (tid < 64) {                               // row softmax of q * d^-0.5
    float vals[64];
    const __bf16* src = &qbf[(size_t)(b * 8192 + n0 + tid) * 512 + h * 64];
    float mx = -3.4e38f;
#pragma unroll
    for (int c8 = 0; c8 < 8; ++c8) {
      bf16x8 v = *(const bf16x8*)&src[c8 * 8];
#pragma unroll
      for (int j = 0; j < 8; ++j) {
        vals[c8 * 8 + j] = (float)v[j];
        mx = fmaxf(mx, vals[c8 * 8 + j]);
      }
    }
    float sum = 0.0f;
#pragma unroll
    for (int e = 0; e < 64; ++e) { vals[e] = __expf(vals[e] - mx); sum += vals[e]; }
    float sc = 0.125f / sum;
#pragma unroll
    for (int c8 = 0; c8 < 8; ++c8) {
      bf16x8 ov;
#pragma unroll
      for (int j = 0; j < 8; ++j) ov[j] = (__bf16)(vals[c8 * 8 + j] * sc);
      *(bf16x8*)&qs[tid * 64 + c8 * 8] = ov;
    }
  }
  __syncthreads();
  floatx8 o[4] = {};
#pragma unroll
  for (int d0 = 0; d0 < 64; d0 += 32) {
    bf16x16 a = a_frag(&cb[(wvid * 16) * 64 + d0], 64, lane);   // rows e
#pragma unroll
    for (int t = 0; t < 4; ++t)
      o[t] = wmma_bf16(a, bT_frag(&qs[(t * 16) * 64 + d0], 64, lane), o[t]);
  }
  const int col = lane & 15, half = lane >> 4;
#pragma unroll
  for (int t = 0; t < 4; ++t) {
    int n = n0 + t * 16 + col;
    float* dstp =
        &attn_out[((size_t)(b * 8 + h) * 8192 + n) * 64 + wvid * 16 + half * 8];
    *(float4*)dstp       = make_float4(o[t][0], o[t][1], o[t][2], o[t][3]);
    *((float4*)dstp + 1) = make_float4(o[t][4], o[t][5], o[t][6], o[t][7]);
  }
}

// ---------------------------------------------------------------------------
// Kernel 7: y^T orientation proj: A = Wproj rows (c), B^T = attn rows (m,
// head-major gather, e-contiguous). Epilogue: bias + float4 stores.
// ---------------------------------------------------------------------------
__global__ __launch_bounds__(128) void proj_kernel(
    const float* __restrict__ attn, const float* __restrict__ Wp,
    const float* __restrict__ bias, float* __restrict__ y) {
  __shared__ __align__(16) __bf16 Ws[64 * 32];
  __shared__ __align__(16) __bf16 As[64 * 32];
  const int m0 = blockIdx.x * 64, c0 = blockIdx.y * 64;
  const int tid = threadIdx.x, lane = tid & 31, wvid = tid >> 5;
  const int srow = tid >> 1, skh = (tid & 1) * 16;
  const int m = m0 + srow, bb = m >> 13, n = m & 8191;
  floatx8 acc[4] = {};
  for (int k0 = 0; k0 < 512; k0 += 32) {
    int kg = k0 + skh, hh = kg >> 6, e = kg & 63;
    stage16(&Wp[(size_t)(c0 + srow) * 512 + k0 + skh], &Ws[srow * 32 + skh]);
    stage16(&attn[((size_t)(bb * 8 + hh) * 8192 + n) * 64 + e], &As[srow * 32 + skh]);
    __syncthreads();
    bf16x16 a = a_frag(&Ws[(wvid * 16) * 32], 32, lane);
#pragma unroll
    for (int t = 0; t < 4; ++t)
      acc[t] = wmma_bf16(a, bT_frag(&As[(t * 16) * 32], 32, lane), acc[t]);
    __syncthreads();
  }
  const int col = lane & 15, half = lane >> 4;
  float bi[8];
#pragma unroll
  for (int i = 0; i < 8; ++i) bi[i] = bias[c0 + wvid * 16 + half * 8 + i];
#pragma unroll
  for (int t = 0; t < 4; ++t) {                 // lane: (c-stripe, m col)
    int mm = m0 + t * 16 + col;
    float* dstp = &y[(size_t)mm * 512 + c0 + wvid * 16 + half * 8];
    *(float4*)dstp = make_float4(acc[t][0] + bi[0], acc[t][1] + bi[1],
                                 acc[t][2] + bi[2], acc[t][3] + bi[3]);
    *((float4*)dstp + 1) = make_float4(acc[t][4] + bi[4], acc[t][5] + bi[5],
                                       acc[t][6] + bi[6], acc[t][7] + bi[7]);
  }
}

// ---------------------------------------------------------------------------
extern "C" void kernel_launch(void* const* d_in, const int* in_sizes, int n_in,
                              void* d_out, int out_size, void* d_ws,
                              size_t ws_size, hipStream_t stream) {
  (void)in_sizes; (void)n_in; (void)out_size; (void)ws_size;
  const float* x   = (const float*)d_in[0];
  const float* Wq  = (const float*)d_in[1];
  const float* Wkv = (const float*)d_in[2];
  const float* Wp  = (const float*)d_in[3];
  const float* bp  = (const float*)d_in[4];

  const size_t BN = 4u * 8192u;                 // 32768 rows
  float* y    = (float*)d_out;                  // (B,N,C)
  float* attn = y + BN * 512;                   // (B,H,N,d)

  __bf16* qbf  = (__bf16*)d_ws;                 // [b][n][c]   32MB
  __bf16* kvbf = qbf + BN * 512;                // [b][n][c]   32MB (k == v)
  __bf16* kvTg = kvbf + BN * 512;               // [b][c][n]   32MB
  float*  ksum = (float*)(kvTg + BN * 512);     // 16 x 64
  __bf16* ctxT = (__bf16*)(ksum + 16 * 64);     // 16 x 64 x 64

  gemm_qkv   <<<dim3(512, 8, 2), 128, 0, stream>>>(x, Wq, Wkv, qbf, kvbf);
  transpose_kv<<<dim3(128, 8, 4), 128, 0, stream>>>(kvbf, kvTg);
  local_attn <<<dim3(128, 4, 4), 128, 0, stream>>>(qbf, kvbf, kvTg, attn);
  ksum_kernel<<<16, 256, 0, stream>>>(kvbf, ksum);
  ctx_kernel <<<16, 128, 0, stream>>>(kvTg, ksum, ctxT);
  linear_out <<<dim3(128, 4, 4), 128, 0, stream>>>(qbf, ctxT, attn);
  proj_kernel<<<dim3(512, 8), 128, 0, stream>>>(attn, Wp, bp, y);
}
